// GeometricFeaturizer_21680994910750
// MI455X (gfx1250) — compile-verified, matrix operand order checked by reading, not run on hardware
//
#include <hip/hip_runtime.h>
#include <hip/hip_bf16.h>
#include <stdint.h>

// ---------------------------------------------------------------------------
// Types for CDNA5 WMMA / TDM intrinsics
// ---------------------------------------------------------------------------
typedef __attribute__((ext_vector_type(16))) __bf16 v16bf;
typedef __attribute__((ext_vector_type(8)))  __bf16 v8bf;
typedef __attribute__((ext_vector_type(8)))  float  v8f;
typedef __attribute__((ext_vector_type(4)))  unsigned int u32x4;
typedef __attribute__((ext_vector_type(8)))  int    i32x8;
typedef __attribute__((ext_vector_type(4)))  int    i32x4;

#if __has_include(<hip/amd_detail/amd_gfx1250_TDM.h>)
#define GF_TDM_6ARG 1
#endif

// ---------------------------------------------------------------------------
// Helpers
// ---------------------------------------------------------------------------
__device__ __forceinline__ float gelu_exact(float x) {
    return 0.5f * x * (1.0f + erff(x * 0.70710678118654752440f));
}

__device__ __forceinline__ void f32_hilo(float x, __bf16& h, __bf16& l) {
    __bf16 hh = (__bf16)x;
    h = hh;
    l = (__bf16)(x - (float)hh);
}

// Build A-fragment (16x32 bf16, hi/lo split) for one lane from an fp32 row.
// lane layout (ISA 7.12.2): lanes 0-15: K = 0..7 & 16..23 ; lanes 16-31: +8.
__device__ __forceinline__ void load_afrag_f32(const float* row, int koff,
                                               v16bf& hi, v16bf& lo) {
    const float4* p0 = (const float4*)(row + koff);
    const float4* p1 = (const float4*)(row + 16 + koff);
    float4 a0 = p0[0], a1 = p0[1], b0 = p1[0], b1 = p1[1];
    float v[16] = {a0.x,a0.y,a0.z,a0.w, a1.x,a1.y,a1.z,a1.w,
                   b0.x,b0.y,b0.z,b0.w, b1.x,b1.y,b1.z,b1.w};
#pragma unroll
    for (int i = 0; i < 16; ++i) { __bf16 h,l; f32_hilo(v[i],h,l); hi[i]=h; lo[i]=l; }
}

// Load a B-fragment from LDS packed as [n][K] bf16 (row base = &buf[n*rowlen + kc]).
__device__ __forceinline__ v16bf load_bfrag(const __bf16* base, int koff) {
    v8bf a = *(const v8bf*)(base + koff);
    v8bf b = *(const v8bf*)(base + 16 + koff);
    v16bf r;
#pragma unroll
    for (int i = 0; i < 8; ++i) { r[i] = a[i]; r[8+i] = b[i]; }
    return r;
}

#define WMMA_BF16(A,B,C) \
    __builtin_amdgcn_wmma_f32_16x16x32_bf16(false,(A),false,(B),(short)0,(C),false,false)

// 3-term hi/lo accumulate: ~fp32-accurate bf16 GEMM step
__device__ __forceinline__ v8f mma_hilo(v16bf ah, v16bf al, v16bf bh, v16bf bl, v8f c) {
    c = WMMA_BF16(ah, bh, c);
    c = WMMA_BF16(ah, bl, c);
    c = WMMA_BF16(al, bh, c);
    return c;
}

__device__ __forceinline__ float wred32(float v) {
    v += __shfl_xor(v, 16, 32); v += __shfl_xor(v, 8, 32);
    v += __shfl_xor(v, 4, 32);  v += __shfl_xor(v, 2, 32);
    v += __shfl_xor(v, 1, 32);
    return v;
}

// ---------------------------------------------------------------------------
// TDM: 2D tile (tile_rows x row_elems fp32, row stride row_stride) -> LDS.
// Descriptor per CDNA5 ISA sec. 8.3/8.4 (D# group0 / group1).
// ---------------------------------------------------------------------------
__device__ __forceinline__ void tdm_load_2d(const float* gsrc, void* ldst,
                                            unsigned tile_rows, unsigned row_elems,
                                            unsigned tensor_rows, unsigned row_stride) {
    uint64_t ga = (uint64_t)(uintptr_t)gsrc;
    unsigned lds = (unsigned)(uintptr_t)ldst;     // low 32 bits of flat addr = LDS offset
    u32x4 g0;
    g0[0] = 1u;                                   // count=1, user descriptor
    g0[1] = lds;                                  // lds_addr
    g0[2] = (unsigned)ga;                         // global_addr[31:0]
    g0[3] = (unsigned)((ga >> 32) & 0x1FFFFFFu) | (2u << 30);  // addr[56:32] | type=2
    i32x8 g1;
    g1[0] = (int)(2u << 16);                                       // data_size=4B
    g1[1] = (int)((row_elems & 0xFFFFu) << 16);                    // tensor_dim0 lo16
    g1[2] = (int)((row_elems >> 16) | ((tensor_rows & 0xFFFFu) << 16)); // dim0 hi / dim1 lo
    g1[3] = (int)((tensor_rows >> 16) | ((row_elems & 0xFFFFu) << 16)); // dim1 hi / tile_dim0
    g1[4] = (int)(tile_rows & 0xFFFFu);                            // tile_dim1, tile_dim2=0
    g1[5] = (int)row_stride;                                       // tensor_dim0_stride lo32
    g1[6] = 0;
    g1[7] = 0;
    i32x4 z4 = {0,0,0,0};
#ifdef GF_TDM_6ARG
    i32x8 z8 = {0,0,0,0,0,0,0,0};
    __builtin_amdgcn_tensor_load_to_lds(g0, g1, z4, z4, z8, 0);
#else
    __builtin_amdgcn_tensor_load_to_lds(g0, g1, z4, z4, 0);
#endif
}

// ---------------------------------------------------------------------------
// Kernel 1: stream x -> Gram(5x5) + pooled ; Jacobi eigh ; SVD features ;
// svd_context = gelu(LN(feats @ svd_w + b)).  One wave per batch row.
// ---------------------------------------------------------------------------
__device__ __forceinline__ void jrot(float A[5][5], float V[5][5], int p, int q) {
    float apq = A[p][q];
    if (fabsf(apq) > 1e-30f) {
        float tau = (A[q][q] - A[p][p]) / (2.0f * apq);
        float tt  = (tau >= 0.0f ? 1.0f : -1.0f) / (fabsf(tau) + sqrtf(1.0f + tau*tau));
        float c   = rsqrtf(1.0f + tt*tt);
        float s   = tt * c;
#pragma unroll
        for (int k = 0; k < 5; ++k) {
            float akp = A[k][p], akq = A[k][q];
            A[k][p] = c*akp - s*akq;  A[k][q] = s*akp + c*akq;
        }
#pragma unroll
        for (int k = 0; k < 5; ++k) {
            float apk = A[p][k], aqk = A[q][k];
            A[p][k] = c*apk - s*aqk;  A[q][k] = s*apk + c*aqk;
        }
#pragma unroll
        for (int k = 0; k < 5; ++k) {
            float vkp = V[k][p], vkq = V[k][q];
            V[k][p] = c*vkp - s*vkq;  V[k][q] = s*vkp + c*vkq;
        }
    }
}

__global__ __launch_bounds__(256)
void gf_k1(const float* __restrict__ x,
           const float* __restrict__ svd_w, const float* __restrict__ svd_b,
           const float* __restrict__ svd_ln_g, const float* __restrict__ svd_ln_b,
           float* __restrict__ pooled, float* __restrict__ out) {
    int t = threadIdx.x, wv = t >> 5, ln = t & 31;
    long long b = (long long)blockIdx.x * 8 + wv;
    const float* xb = x + b * 2560;

    float gg[15];
#pragma unroll
    for (int j = 0; j < 15; ++j) gg[j] = 0.0f;

    for (int it = 0; it < 16; ++it) {
        int d = ln + (it << 5);
        __builtin_prefetch(xb + d + 256, 0, 0);   // global_prefetch
        float xs[5];
#pragma unroll
        for (int s = 0; s < 5; ++s) xs[s] = xb[s * 512 + d];
        pooled[b * 512 + d] = (xs[0]+xs[1]+xs[2]+xs[3]+xs[4]) * 0.2f;
        int q = 0;
#pragma unroll
        for (int s = 0; s < 5; ++s)
#pragma unroll
            for (int u = s; u < 5; ++u) { gg[q] = fmaf(xs[s], xs[u], gg[q]); ++q; }
    }
#pragma unroll
    for (int j = 0; j < 15; ++j) gg[j] = wred32(gg[j]);

    float A[5][5], V[5][5];
    { int q = 0;
#pragma unroll
      for (int s = 0; s < 5; ++s)
#pragma unroll
        for (int u = s; u < 5; ++u) { A[s][u] = gg[q]; A[u][s] = gg[q]; ++q; } }
#pragma unroll
    for (int i = 0; i < 5; ++i)
#pragma unroll
        for (int j = 0; j < 5; ++j) V[i][j] = (i == j) ? 1.0f : 0.0f;

#pragma unroll
    for (int sw = 0; sw < 8; ++sw)
#pragma unroll
        for (int p = 0; p < 4; ++p)
#pragma unroll
            for (int q = p + 1; q < 5; ++q) jrot(A, V, p, q);

    // deterministic sign convention: largest-|component| of each eigvec >= 0
#pragma unroll
    for (int c = 0; c < 5; ++c) {
        float best = V[0][c]; float ab = fabsf(best);
#pragma unroll
        for (int r = 1; r < 5; ++r) { float v = V[r][c]; if (fabsf(v) > ab) { ab = fabsf(v); best = v; } }
        if (best < 0.0f) {
#pragma unroll
            for (int r = 0; r < 5; ++r) V[r][c] = -V[r][c];
        }
    }

    float ev[5]; int ci[5];
#pragma unroll
    for (int i = 0; i < 5; ++i) { ev[i] = A[i][i]; ci[i] = i; }
#pragma unroll
    for (int i = 0; i < 4; ++i)
#pragma unroll
        for (int j = i + 1; j < 5; ++j)
            if (ev[j] > ev[i]) { float te = ev[i]; ev[i] = ev[j]; ev[j] = te;
                                 int tc = ci[i]; ci[i] = ci[j]; ci[j] = tc; }

    float sv[5], ssum = 0.0f;
#pragma unroll
    for (int i = 0; i < 5; ++i) {
        float s = sqrtf(fmaxf(ev[i], 0.0f));
        s = fmaxf(s, 1e-6f);
        if (!isfinite(s)) s = 1.0f;
        sv[i] = s; ssum += s;
    }
    float feats[12];
#pragma unroll
    for (int i = 0; i < 5; ++i) feats[i] = sv[i] / (ssum + 1e-8f);
    float dsum = 0.0f;
#pragma unroll
    for (int i = 0; i < 5; ++i) {
        int c = ci[i];
        float v = (c==0)?V[i][0]:(c==1)?V[i][1]:(c==2)?V[i][2]:(c==3)?V[i][3]:V[i][4];
        if (!isfinite(v)) v = 0.0f;
        feats[5 + i] = v; dsum += v * v;
    }
    float tot = 0.0f;
#pragma unroll
    for (int r = 0; r < 5; ++r)
#pragma unroll
        for (int c = 0; c < 5; ++c) { float v = V[r][c]; if (!isfinite(v)) v = 0.0f; tot += v*v; }
    feats[10] = fmaxf(tot - dsum, 0.0f);
    float ent = 0.0f;
#pragma unroll
    for (int i = 0; i < 5; ++i) ent -= feats[i] * logf(fmaxf(feats[i], 1e-8f));
    feats[11] = ent;
#pragma unroll
    for (int f = 0; f < 12; ++f) if (!isfinite(feats[f])) feats[f] = 0.0f;

    // feats(12) @ svd_w(12,256) + bias, LN over 256, GELU
    float y[8];
#pragma unroll
    for (int i = 0; i < 8; ++i) {
        int n = ln + i * 32;
        float acc = svd_b[n];
#pragma unroll
        for (int f = 0; f < 12; ++f) acc = fmaf(feats[f], svd_w[f * 256 + n], acc);
        y[i] = acc;
    }
    float s = 0.0f, s2 = 0.0f;
#pragma unroll
    for (int i = 0; i < 8; ++i) { s += y[i]; s2 += y[i]*y[i]; }
    s = wred32(s); s2 = wred32(s2);
    float mean = s * (1.0f/256.0f);
    float var  = s2 * (1.0f/256.0f) - mean * mean;
    float rstd = rsqrtf(var + 1e-5f);
#pragma unroll
    for (int i = 0; i < 8; ++i) {
        int n = ln + i * 32;
        float v = (y[i] - mean) * rstd * svd_ln_g[n] + svd_ln_b[n];
        out[b * 768 + n] = gelu_exact(v);
    }
}

// ---------------------------------------------------------------------------
// Kernel 2: emb_pre = pooled(B,512) @ emb_w(512,256) + emb_b
// TDM double-buffered weight tiles -> LDS, bf16 hi/lo WMMA, 8 waves x 16 rows.
// ---------------------------------------------------------------------------
__global__ __launch_bounds__(256)
void gf_k2(const float* __restrict__ pooled, const float* __restrict__ emb_w,
           const float* __restrict__ emb_b, float* __restrict__ emb_pre) {
    extern __shared__ unsigned char smemraw[];
    float*  stage0 = (float*)smemraw;            // 32x256 f32
    float*  stage1 = stage0 + 8192;              // 32x256 f32
    __bf16* sBHi   = (__bf16*)(stage1 + 8192);   // [n][kk] packed, 256x32
    __bf16* sBLo   = sBHi + 8192;
    float*  sBias  = (float*)(sBLo + 8192);      // 256

    int t = threadIdx.x, wv = t >> 5, ln = t & 31;
    int am = ln & 15, koff = (ln >> 4) * 8;
    if (t < 256) sBias[t] = emb_b[t];
    long long rbase = ((long long)blockIdx.x * 8 + wv) * 16;

    v8f acc[16];
#pragma unroll
    for (int i = 0; i < 16; ++i)
#pragma unroll
        for (int j = 0; j < 8; ++j) acc[i][j] = 0.0f;

    if (wv == 0) tdm_load_2d(emb_w, stage0, 32, 256, 512, 256);

    int kb = 0;
    for (int kc = 0; kc < 512; kc += 32, kb ^= 1) {
        if (wv == 0) {
            if (kc + 32 < 512) {
                tdm_load_2d(emb_w + (size_t)(kc + 32) * 256, kb ? stage0 : stage1,
                            32, 256, 512, 256);
                __builtin_amdgcn_s_wait_tensorcnt(1);
            } else {
                __builtin_amdgcn_s_wait_tensorcnt(0);
            }
        }
        __syncthreads();
        // convert staged f32 [kk][n] -> packed bf16 hi/lo [n][kk]
        float* st = kb ? stage1 : stage0;
        for (int i = t; i < 8192; i += 256) {
            int n = i >> 5, kk = i & 31;
            __bf16 h, l; f32_hilo(st[kk * 256 + n], h, l);
            sBHi[i] = h; sBLo[i] = l;
        }
        __syncthreads();

        const float* arow = pooled + (size_t)(rbase + am) * 512 + kc;
        v16bf ahi, alo; load_afrag_f32(arow, koff, ahi, alo);
#pragma unroll
        for (int nt = 0; nt < 16; ++nt) {
            int n = nt * 16 + am;
            v16bf bhi = load_bfrag(sBHi + n * 32, koff);
            v16bf blo = load_bfrag(sBLo + n * 32, koff);
            acc[nt] = mma_hilo(ahi, alo, bhi, blo, acc[nt]);
        }
        __syncthreads();
    }

    // epilogue: + bias, store (C layout: lane n = am, row = 8*(ln>>4)+v)
    long long r0 = rbase + (ln >> 4) * 8;
#pragma unroll
    for (int nt = 0; nt < 16; ++nt) {
        int col = nt * 16 + am;
        float bb = sBias[col];
#pragma unroll
        for (int v = 0; v < 8; ++v)
            emb_pre[(size_t)(r0 + v) * 256 + col] = acc[nt][v] + bb;
    }
}

// ---------------------------------------------------------------------------
// Kernel 3: per 16-row tile: LN+L2 -> emb ; cos vs anchors (WMMA) ; top-3 gate;
// compartment MLPs (WMMA) ; per-compartment LN ; write pw + emb slices.
// ---------------------------------------------------------------------------
__global__ __launch_bounds__(256)
void gf_k3(const float* __restrict__ emb_pre,
           const float* __restrict__ emb_ln_g, const float* __restrict__ emb_ln_b,
           const float* __restrict__ anchors,
           const float* __restrict__ w1, const float* __restrict__ b1,
           const float* __restrict__ w2, const float* __restrict__ b2,
           const float* __restrict__ lng, const float* __restrict__ lnb,
           float* __restrict__ out) {
    extern __shared__ unsigned char smemraw[];
    __bf16* sAncHi = (__bf16*)smemraw;       // [a][k] 32x256
    __bf16* sAncLo = sAncHi + 8192;
    __bf16* sW1Hi  = sAncLo + 8192;          // [col(0..511)][a] col = c*64+j
    __bf16* sW1Lo  = sW1Hi + 16384;
    __bf16* sW2Hi  = sW1Lo + 16384;          // [(c*32+j)][k(0..63)]
    __bf16* sW2Lo  = sW2Hi + 16384;
    float*  sE     = (float*)(sW2Lo + 16384);  // 8 waves x 16x256 f32
    float*  sAn    = sE + 8 * 4096;            // 32
    float*  sLg    = sAn + 32;                 // 256
    float*  sLb    = sLg + 256;                // 256

    int t = threadIdx.x, wv = t >> 5, ln = t & 31;
    int am = ln & 15, koff = (ln >> 4) * 8;

    // ---- block prep: anchor norms, LN params, packed weights ----
    if (t < 32) {
        float s = 0.0f;
        for (int k = 0; k < 256; ++k) { float v = anchors[t * 256 + k]; s = fmaf(v, v, s); }
        sAn[t] = fmaxf(sqrtf(s), 1e-12f);
    }
    sLg[t] = emb_ln_g[t]; sLb[t] = emb_ln_b[t];
    __syncthreads();
    for (int i = t; i < 8192; i += 256) {
        int a = i >> 8;
        __bf16 h, l; f32_hilo(anchors[i] / sAn[a], h, l);
        sAncHi[i] = h; sAncLo[i] = l;
    }
    for (int i = t; i < 16384; i += 256) {       // comp_w1[c][a][j] -> [c*64+j][a]
        int c = i >> 11, a = (i >> 6) & 31, j = i & 63;
        int dst = (c * 64 + j) * 32 + a;
        __bf16 h, l; f32_hilo(w1[i], h, l);
        sW1Hi[dst] = h; sW1Lo[dst] = l;
    }
    for (int i = t; i < 16384; i += 256) {       // comp_w2[c][k][j] -> [c*32+j][k]
        int c = i >> 11, k = (i >> 5) & 63, j = i & 31;
        int dst = (c * 32 + j) * 64 + k;
        __bf16 h, l; f32_hilo(w2[i], h, l);
        sW2Hi[dst] = h; sW2Lo[dst] = l;
    }
    __syncthreads();

    float* E = sE + wv * 4096;
    long long rbase = ((long long)blockIdx.x * 8 + wv) * 16;

    // ---- load 16x256 emb_pre tile ----
    for (int r = 0; r < 16; ++r)
#pragma unroll
        for (int i = 0; i < 8; ++i) {
            int cc = ln + i * 32;
            E[r * 256 + cc] = emb_pre[(size_t)(rbase + r) * 256 + cc];
        }

    // ---- LN + L2 normalize (2 lanes per row) ----
    {
        int r = am, base = (ln >> 4) * 128;
        float s = 0.0f, s2 = 0.0f;
        for (int i = 0; i < 128; ++i) { float v = E[r * 256 + base + i]; s += v; s2 = fmaf(v, v, s2); }
        s += __shfl_xor(s, 16, 32); s2 += __shfl_xor(s2, 16, 32);
        float mean = s * (1.0f/256.0f);
        float var  = s2 * (1.0f/256.0f) - mean * mean;
        float rstd = rsqrtf(var + 1e-5f);
        float n2 = 0.0f;
        for (int i = 0; i < 128; ++i) {
            int cc = base + i;
            float y = (E[r * 256 + cc] - mean) * rstd * sLg[cc] + sLb[cc];
            E[r * 256 + cc] = y; n2 = fmaf(y, y, n2);
        }
        n2 += __shfl_xor(n2, 16, 32);
        float rn = 1.0f / fmaxf(sqrtf(n2), 1e-12f);
        for (int i = 0; i < 128; ++i) E[r * 256 + base + i] *= rn;
    }
    // store emb slice (coalesced)
    for (int r = 0; r < 16; ++r)
#pragma unroll
        for (int i = 0; i < 8; ++i) {
            int cc = ln + i * 32;
            out[(size_t)(rbase + r) * 768 + 512 + cc] = E[r * 256 + cc];
        }

    // ---- cos = emb(16x256) @ ancN^T(256x32) via WMMA hi/lo ----
    v8f ca0, ca1;
#pragma unroll
    for (int j = 0; j < 8; ++j) { ca0[j] = 0.0f; ca1[j] = 0.0f; }
    for (int kc = 0; kc < 256; kc += 32) {
        v16bf ahi, alo; load_afrag_f32(E + am * 256 + kc, koff, ahi, alo);
        v16bf b0h = load_bfrag(sAncHi + (am)      * 256 + kc, koff);
        v16bf b0l = load_bfrag(sAncLo + (am)      * 256 + kc, koff);
        v16bf b1h = load_bfrag(sAncHi + (16 + am) * 256 + kc, koff);
        v16bf b1l = load_bfrag(sAncLo + (16 + am) * 256 + kc, koff);
        ca0 = mma_hilo(ahi, alo, b0h, b0l, ca0);
        ca1 = mma_hilo(ahi, alo, b1h, b1l, ca1);
    }
    // dump cos tile (reuse E region as scratch from here on)
    float* sc  = E;           // [0..511]   cos 16x32
    float* str = E + 512;     // [512..1023] tri_gated 16x32
    float* ht  = E + 1024;    // [1024..2047] h tile 16x64
    float* ot  = E + 2048;    // [2048..2559] out tile 16x32
    {
        int rr = (ln >> 4) * 8;
#pragma unroll
        for (int v = 0; v < 8; ++v) {
            sc[(rr + v) * 32 + am]      = ca0[v];
            sc[(rr + v) * 32 + 16 + am] = ca1[v];
        }
    }
    // ---- top-3 gate, tri_gated (lanes 0-15, one row each) ----
    if (ln < 16) {
        int r = ln;
        float cv[32];
#pragma unroll
        for (int a = 0; a < 32; ++a) cv[a] = sc[r * 32 + a];
        float m1 = -3.4e38f, m2 = -3.4e38f, m3 = -3.4e38f;
#pragma unroll
        for (int a = 0; a < 32; ++a) {
            float v = cv[a];
            if (v > m1)      { m3 = m2; m2 = m1; m1 = v; }
            else if (v > m2) { m3 = m2; m2 = v; }
            else if (v > m3) { m3 = v; }
        }
        float se = 0.0f;
#pragma unroll
        for (int a = 0; a < 32; ++a) if (cv[a] >= m3) se += expf(cv[a] - m1);
        float inv = 1.0f / se;
#pragma unroll
        for (int a = 0; a < 32; ++a) {
            float g = (cv[a] >= m3) ? expf(cv[a] - m1) * inv : 0.0f;
            str[r * 32 + a] = (1.0f - cv[a]) * g;
        }
    }
    // tri A-fragment (K=32, single chunk)
    v16bf thi, tlo; load_afrag_f32(str + am * 32, 0 + koff, thi, tlo);

    // ---- compartments ----
    for (int c = 0; c < 8; ++c) {
        // h = gelu(tri @ W1[c] + b1[c])  : 4 N-tiles of 16
        v8f ha[4];
#pragma unroll
        for (int nt = 0; nt < 4; ++nt) {
#pragma unroll
            for (int j = 0; j < 8; ++j) ha[nt][j] = 0.0f;
            int coln = c * 64 + nt * 16 + am;
            v16bf bh = load_bfrag(sW1Hi + coln * 32, koff);
            v16bf bl = load_bfrag(sW1Lo + coln * 32, koff);
            ha[nt] = mma_hilo(thi, tlo, bh, bl, ha[nt]);
        }
        int rr = (ln >> 4) * 8;
#pragma unroll
        for (int nt = 0; nt < 4; ++nt) {
            int jj = nt * 16 + am;
            float bb = b1[c * 64 + jj];
#pragma unroll
            for (int v = 0; v < 8; ++v)
                ht[(rr + v) * 64 + jj] = gelu_exact(ha[nt][v] + bb);
        }
        // out = h @ W2[c] + b2[c] : 2 N-tiles, K=64 (2 chunks)
        v8f oa[2];
#pragma unroll
        for (int j = 0; j < 8; ++j) { oa[0][j] = 0.0f; oa[1][j] = 0.0f; }
#pragma unroll
        for (int kc2 = 0; kc2 < 64; kc2 += 32) {
            v16bf ah2, al2; load_afrag_f32(ht + am * 64 + kc2, koff, ah2, al2);
#pragma unroll
            for (int nt3 = 0; nt3 < 2; ++nt3) {
                int coln = c * 32 + nt3 * 16 + am;
                v16bf bh = load_bfrag(sW2Hi + coln * 64 + kc2, koff);
                v16bf bl = load_bfrag(sW2Lo + coln * 64 + kc2, koff);
                oa[nt3] = mma_hilo(ah2, al2, bh, bl, oa[nt3]);
            }
        }
#pragma unroll
        for (int nt3 = 0; nt3 < 2; ++nt3) {
            int jj = nt3 * 16 + am;
            float bb = b2[c * 32 + jj];
#pragma unroll
            for (int v = 0; v < 8; ++v)
                ot[(rr + v) * 32 + jj] = oa[nt3][v] + bb;
        }
        // per-row LN over H=32, write pw slice
        if (ln < 16) {
            int r = ln;
            float s = 0.0f, s2 = 0.0f;
#pragma unroll
            for (int j = 0; j < 32; ++j) { float v = ot[r * 32 + j]; s += v; s2 = fmaf(v, v, s2); }
            float mean = s * (1.0f/32.0f);
            float var  = s2 * (1.0f/32.0f) - mean * mean;
            float rstd = rsqrtf(var + 1e-5f);
#pragma unroll
            for (int j = 0; j < 32; ++j) {
                float v = (ot[r * 32 + j] - mean) * rstd * lng[c * 32 + j] + lnb[c * 32 + j];
                out[(size_t)(rbase + r) * 768 + 256 + c * 32 + j] = v;
            }
        }
    }
}

// ---------------------------------------------------------------------------
// Launch
// ---------------------------------------------------------------------------
extern "C" void kernel_launch(void* const* d_in, const int* in_sizes, int n_in,
                              void* d_out, int out_size, void* d_ws, size_t ws_size,
                              hipStream_t stream) {
    const float* x        = (const float*)d_in[0];
    const float* svd_w    = (const float*)d_in[1];
    const float* svd_b    = (const float*)d_in[2];
    const float* svd_ln_g = (const float*)d_in[3];
    const float* svd_ln_b = (const float*)d_in[4];
    const float* emb_w    = (const float*)d_in[5];
    const float* emb_b    = (const float*)d_in[6];
    const float* emb_ln_g = (const float*)d_in[7];
    const float* emb_ln_b = (const float*)d_in[8];
    const float* anchors  = (const float*)d_in[9];
    const float* comp_w1  = (const float*)d_in[10];
    const float* comp_b1  = (const float*)d_in[11];
    const float* comp_w2  = (const float*)d_in[12];
    const float* comp_b2  = (const float*)d_in[13];
    const float* comp_lng = (const float*)d_in[14];
    const float* comp_lnb = (const float*)d_in[15];
    float* out = (float*)d_out;

    int B = in_sizes[0] / (5 * 512);
    float* pooled  = (float*)d_ws;                       // B x 512 f32
    float* emb_pre = pooled + (size_t)B * 512;           // B x 256 f32

    gf_k1<<<B / 8, 256, 0, stream>>>(x, svd_w, svd_b, svd_ln_g, svd_ln_b, pooled, out);

    size_t k2smem = (size_t)(2 * 8192 + 256) * sizeof(float) + (size_t)2 * 8192 * sizeof(__bf16);
    gf_k2<<<B / 128, 256, k2smem, stream>>>(pooled, emb_w, emb_b, emb_pre);

    size_t k3smem = (size_t)(2 * 8192 + 4 * 16384) * sizeof(__bf16)
                  + (size_t)(8 * 4096 + 32 + 512) * sizeof(float);
    gf_k3<<<B / 128, 256, k3smem, stream>>>(emb_pre, emb_ln_g, emb_ln_b, anchors,
                                            comp_w1, comp_b1, comp_w2, comp_b2,
                                            comp_lng, comp_lnb, out);
}
